// ScaledDotProductMHA_13649406067532
// MI455X (gfx1250) — matrix operand order repaired and checked by reading, not compiled, not run on hardware
//
#include <hip/hip_runtime.h>

typedef __bf16 bf16;
typedef __attribute__((ext_vector_type(16))) __bf16 v16bf;
typedef __attribute__((ext_vector_type(8)))  __bf16 v8bf;
typedef __attribute__((ext_vector_type(8)))  float   v8f;

namespace {

constexpr int kN    = 2048;  // sequence length
constexpr int kD    = 64;    // head dim
constexpr int kBH   = 32;    // B*H
constexpr int kBlkQ = 128;   // q rows per workgroup (8 waves x 16)
constexpr int kBlkK = 64;    // kv keys per staged block (4 key tiles)

// scale = 1/sqrt(64), with log2(e) folded in so all exponentials are exp2.
constexpr float kScaleLog2e = 0.125f * 1.44269504088896340736f;
constexpr float kNegBig     = -1e30f;

union F8 { v8f v; float e[8]; };
union A16 {
  v16bf v;
  bf16  e[16];
  struct { v8bf lo, hi; } h;
};
union Ones { v16bf v; unsigned int u[8]; };
union BF2  { unsigned int u; bf16 h[2]; };

__device__ __forceinline__ unsigned int pk2(float a, float b) {
  BF2 t; t.h[0] = (bf16)a; t.h[1] = (bf16)b; return t.u;
}

__device__ __forceinline__ float f4get(const float4& v, int j) {
  const float a[4] = {v.x, v.y, v.z, v.w};
  return a[j];
}

__device__ __forceinline__ v8f wmma_bf16(v16bf a, v16bf b, v8f c) {
  // D(16x16 f32) = A(16x32 bf16) * B(32x16 bf16) + C
  return __builtin_amdgcn_wmma_f32_16x16x32_bf16(
      /*neg_a=*/false, a, /*neg_b=*/false, b,
      /*c_mod=*/(short)0, c, /*reuse_a=*/false, /*reuse_b=*/false);
}

// XOR-butterfly max across the 16 lanes holding one score row, using
// ds_swizzle (group-of-32 mode: and=0x1f, or=0, xor=MASK).
template <int MASK>
__device__ __forceinline__ float max_swz(float x) {
  int o = __builtin_amdgcn_ds_swizzle(__float_as_int(x), (MASK << 10) | 0x1f);
  return fmaxf(x, __int_as_float(o));
}
__device__ __forceinline__ float rowmax16(float x) {
  x = max_swz<1>(x);
  x = max_swz<2>(x);
  x = max_swz<4>(x);
  x = max_swz<8>(x);
  return x;
}

} // namespace

__global__ __launch_bounds__(256)
void fa_causal_bf16_wmma(const float* __restrict__ Kg,   // keys    [BH][N][64]
                         const float* __restrict__ Qg,   // queries [BH][N][64]
                         const float* __restrict__ Vg,   // values  [BH][N][64]
                         float* __restrict__ Og)         // out     [BH][N][64]
{
  __shared__ bf16 Kl[2][kBlkK][kD];     // double-buffered K  (16 KB)
  __shared__ bf16 Vt[2][kD][kBlkK];     // double-buffered V^T (16 KB)
  __shared__ bf16 Pl[8][16][kBlkK];     // per-wave P staging (16 KB)

  const int tid  = threadIdx.x;
  const int wave = tid >> 5;
  const int lane = tid & 31;
  const int half = lane >> 4;     // K-half / row-half selector
  const int ln   = lane & 15;     // matrix row (A) or column (B/C/D)

  const int bh = blockIdx.y;                 // fused batch*head
  const int qb = blockIdx.x * kBlkQ;         // block's first q row
  const int qw = qb + wave * 16;             // this wave's first q row

  // Staging unit: each thread owns a 4-key x 4-d patch.
  const int kg = (tid >> 4) << 2;            // first key of patch (0..60)
  const int dg = (tid & 15) << 2;            // first d of patch   (0..60)

  const float* __restrict__ Kbase = Kg + (size_t)bh * kN * kD;
  const float* __restrict__ Vbase = Vg + (size_t)bh * kN * kD;
  const float* __restrict__ Qrow  = Qg + ((size_t)bh * kN + qw + ln) * kD;

  // ---- Load Q tile (16x64) into WMMA 16-bit A layout, scale folded in ----
  A16 qa[2];
#pragma unroll
  for (int c = 0; c < 2; ++c) {
#pragma unroll
    for (int e = 0; e < 16; ++e) {
      const int k = half * 16 + e;
      qa[c].e[e] = (bf16)(Qrow[c * 32 + k] * kScaleLog2e);
    }
  }

  // All-ones B operand for the row-sum-by-WMMA trick.
  Ones ones;
#pragma unroll
  for (int i = 0; i < 8; ++i) ones.u[i] = 0x3F803F80u;  // bf16 1.0 x2

  // ---- Flash-attention state (rows r+half*8, matching C/D layout) ----
  F8 acc[4];            // O accumulator, 16x64
  F8 accS0, accS1;      // independent row-sum accumulators (ones-column WMMA)
  F8 rmax;
#pragma unroll
  for (int t = 0; t < 4; ++t)
#pragma unroll
    for (int r = 0; r < 8; ++r) acc[t].e[r] = 0.0f;
#pragma unroll
  for (int r = 0; r < 8; ++r) {
    accS0.e[r] = 0.0f; accS1.e[r] = 0.0f; rmax.e[r] = kNegBig;
  }

  const int nblk = (qb + kBlkQ) / kBlkK;  // uniform over the workgroup

  // ---- Pipeline helpers (4-key x 4-d patch per thread) ----
  float4 kr[4], vr[4];   // in-flight global data, kept raw so the wait is late
  auto load_block = [&](int j) {
    const float* kp0 = Kbase + (size_t)(j * kBlkK + kg) * kD + dg;
    const float* vp0 = Vbase + (size_t)(j * kBlkK + kg) * kD + dg;
#pragma unroll
    for (int i = 0; i < 4; ++i) {
      kr[i] = *(const float4*)(kp0 + (size_t)i * kD);
      vr[i] = *(const float4*)(vp0 + (size_t)i * kD);
    }
  };
  auto store_block = [&](int buf) {
#pragma unroll
    for (int i = 0; i < 4; ++i) {      // K: row kg+i, cols dg..dg+3
      uint2 w;
      w.x = pk2(kr[i].x, kr[i].y);
      w.y = pk2(kr[i].z, kr[i].w);
      *(uint2*)&Kl[buf][kg + i][dg] = w;
    }
#pragma unroll
    for (int jj = 0; jj < 4; ++jj) {   // V^T: row dg+jj, cols kg..kg+3
      uint2 w;
      w.x = pk2(f4get(vr[0], jj), f4get(vr[1], jj));
      w.y = pk2(f4get(vr[2], jj), f4get(vr[3], jj));
      *(uint2*)&Vt[buf][dg + jj][kg] = w;
    }
  };

  // Prologue: stage block 0.
  load_block(0);
  store_block(0);
  __syncthreads();

  for (int j = 0; j < nblk; ++j) {
    const int kv0 = j * kBlkK;
    const int buf = j & 1;

    // Issue next block's global loads now; compute below hides their latency.
    if (j + 1 < nblk) load_block(j + 1);

    if (kv0 <= qw + 15) {   // uniform per wave (causal range)
      // ---- S = Q * K^T : four 16x16 key tiles ----
      F8 s[4];
#pragma unroll
      for (int kt = 0; kt < 4; ++kt) {
        v8f acc_s = {};
        v16bf b0 = *(const v16bf*)&Kl[buf][kt * 16 + ln][half * 16];
        acc_s = wmma_bf16(qa[0].v, b0, acc_s);
        v16bf b1 = *(const v16bf*)&Kl[buf][kt * 16 + ln][32 + half * 16];
        acc_s = wmma_bf16(qa[1].v, b1, acc_s);
        s[kt].v = acc_s;
      }

      // ---- causal mask: only the single diagonal block needs it ----
      if (kv0 + kBlkK - 1 > qw) {
#pragma unroll
        for (int r = 0; r < 8; ++r) {
          const int qrow = qw + r + half * 8;
#pragma unroll
          for (int kt = 0; kt < 4; ++kt) {
            if (kv0 + kt * 16 + ln > qrow) s[kt].e[r] = kNegBig;
          }
        }
      }

      // ---- online softmax (base-2; row = r + half*8, col = ln) ----
      F8 corr;
#pragma unroll
      for (int r = 0; r < 8; ++r) {
        float m = fmaxf(fmaxf(s[0].e[r], s[1].e[r]),
                        fmaxf(s[2].e[r], s[3].e[r]));
        m = rowmax16(m);
        const float nm = fmaxf(rmax.e[r], m);
        const float cr = __builtin_amdgcn_exp2f(rmax.e[r] - nm);
        rmax.e[r] = nm;
        corr.e[r] = cr;
#pragma unroll
        for (int kt = 0; kt < 4; ++kt)
          s[kt].e[r] = __builtin_amdgcn_exp2f(s[kt].e[r] - nm);
      }

      // rescale running accumulators (O tiles + row-sum tiles)
#pragma unroll
      for (int t = 0; t < 4; ++t)
#pragma unroll
        for (int r = 0; r < 8; ++r) acc[t].e[r] *= corr.e[r];
#pragma unroll
      for (int r = 0; r < 8; ++r) {
        accS0.e[r] *= corr.e[r];
        accS1.e[r] *= corr.e[r];
      }

      // ---- C-layout -> A-layout for P via per-wave LDS staging ----
#pragma unroll
      for (int r = 0; r < 8; ++r)
#pragma unroll
        for (int kt = 0; kt < 4; ++kt)
          Pl[wave][r + half * 8][kt * 16 + ln] = (bf16)s[kt].e[r];

      A16 pa[2];  // P as A operand, two 16x32 K-chunks (keys 0..31, 32..63)
#pragma unroll
      for (int c = 0; c < 2; ++c) {
        pa[c].h.lo = *(const v8bf*)&Pl[wave][ln][c * 32 + half * 16];
        pa[c].h.hi = *(const v8bf*)&Pl[wave][ln][c * 32 + half * 16 + 8];
      }

      // ---- O += P * V ; rowsum += P * ones (independent chains) ----
#pragma unroll
      for (int t = 0; t < 4; ++t) {
#pragma unroll
        for (int c = 0; c < 2; ++c) {
          v16bf b = *(const v16bf*)&Vt[buf][t * 16 + ln][c * 32 + half * 16];
          acc[t].v = wmma_bf16(pa[c].v, b, acc[t].v);
        }
      }
      accS0.v = wmma_bf16(pa[0].v, ones.v, accS0.v);
      accS1.v = wmma_bf16(pa[1].v, ones.v, accS1.v);
    }

    // Stage next block into the other buffer (s_wait_loadcnt lands here,
    // after the compute above has covered the latency).
    if (j + 1 < nblk) store_block(buf ^ 1);
    __syncthreads();
  }

  // ---- Epilogue: O *= 1/rowsum (rcp; denominators >= 1), write fp32 ----
  float* __restrict__ Orow = Og + ((size_t)bh * kN + qw) * kD;
  F8 inv;
#pragma unroll
  for (int r = 0; r < 8; ++r)
    inv.e[r] = __builtin_amdgcn_rcpf(accS0.e[r] + accS1.e[r]);
#pragma unroll
  for (int t = 0; t < 4; ++t) {
#pragma unroll
    for (int r = 0; r < 8; ++r) {
      Orow[(size_t)(r + half * 8) * kD + t * 16 + ln] =
          acc[t].e[r] * inv.e[r];
    }
  }
}

extern "C" void kernel_launch(void* const* d_in, const int* in_sizes, int n_in,
                              void* d_out, int out_size, void* d_ws, size_t ws_size,
                              hipStream_t stream) {
  // setup_inputs order: keys, queries, values (all fp32, [B][N][H][D] == [BH][N][D])
  const float* keys    = (const float*)d_in[0];
  const float* queries = (const float*)d_in[1];
  const float* values  = (const float*)d_in[2];
  float* out = (float*)d_out;

  dim3 grid(kN / kBlkQ, kBH);   // (16, 32)
  dim3 block(256);              // 8 waves
  fa_causal_bf16_wmma<<<grid, block, 0, stream>>>(keys, queries, values, out);
}
